// IntraVoxelConv_8375186227639
// MI455X (gfx1250) — matrix operand order repaired and checked by reading, not compiled, not run on hardware
//
#include <hip/hip_runtime.h>

typedef __attribute__((ext_vector_type(16))) _Float16 v16h;
typedef __attribute__((ext_vector_type(8)))  float    v8f;

#define CIN   16
#define COUT  16
#define DIM   96
#define TILE  4          // d,h tile per workgroup
#define HALO  6          // TILE + 2
#define WROW  98         // 96 + 2 halo in w
#define NPAIR 14         // 28 taps (27 real + 1 zero pad) / 2 per WMMA

#define LX_HALVES (HALO*HALO*WROW*CIN)   // 56448 halves = 112896 B
#define WL_HALVES (NPAIR*32*16)          // 7168 halves  = 14336 B
#define LM_FLOATS (TILE*TILE*DIM)        // 1536 floats  = 6144 B
#define LDS_BYTES (LX_HALVES*2 + WL_HALVES*2 + LM_FLOATS*4 + 16*4)

__global__ __launch_bounds__(256)
void mvc_wmma_kernel(const float* __restrict__ x,
                     const float* __restrict__ w,
                     float* __restrict__ out) {
  extern __shared__ char smem[];
  _Float16* lx   = (_Float16*)smem;                 // log(x) halo tile [z][y][x][ci], f16
  _Float16* wlds = lx + LX_HALVES;                  // per-lane pre-swizzled A fragments
  float*    logM = (float*)(wlds + WL_HALVES);      // per-voxel base point (core voxels)
  float*    wadj = logM + LM_FLOATS;                // (1 - sum(w)) per cout

  const int tid = threadIdx.x;
  const int yt  = blockIdx.x;      // 0..23
  const int zt  = blockIdx.y;      // 0..23
  const int b   = blockIdx.z;      // 0..1
  const int z0  = zt*TILE - 1;
  const int y0  = yt*TILE - 1;

  // ---- Stage log(x) into LDS (f16), zero-padded halo; coalesced along w ----
  for (int e = tid; e < LX_HALVES; e += 256) {
    int xh    = e % WROW;          // fastest: contiguous global reads
    int rest  = e / WROW;
    int dy    = rest % HALO;
    int rest2 = rest / HALO;
    int dz    = rest2 % HALO;
    int ci    = rest2 / HALO;
    int zz = z0 + dz, yy = y0 + dy, xx = xh - 1;
    float v = 0.0f;
    if ((unsigned)zz < DIM && (unsigned)yy < DIM && (unsigned)xx < DIM)
      v = __logf(x[(((b*CIN + ci)*DIM + zz)*DIM + yy)*DIM + xx]);
    lx[((dz*HALO + dy)*WROW + xh)*CIN + ci] = (_Float16)v;
  }

  // ---- Stage weights pre-swizzled into WMMA A-fragment per-lane layout ----
  // K = tap_in_pair*16 + ci. Lanes 0-15: halves 0..15 -> K 0..15.
  // Lanes 16-31: halves 0..15 -> K 16..31.  M = lane&15 = cout.
  for (int e = tid; e < WL_HALVES; e += 256) {
    int h    = e & 15;
    int lane = (e >> 4) & 31;
    int pair = e >> 9;
    int co   = lane & 15;
    int cib  = (lane >= 16) ? 8 : 0;
    int tap  = pair*2 + (cib ? 1 : 0);
    int ci   = h;
    float wv = (tap < 27) ? w[co*432 + tap*16 + ci] : 0.0f;  // tap 27 = zero pad
    wlds[e] = (_Float16)wv;
  }
  if (tid < 16) {
    float s = 0.0f;
    for (int i = 0; i < 432; ++i) s += w[tid*432 + i];
    wadj[tid] = 1.0f - s;
  }
  __syncthreads();

  // ---- Per-voxel base point: logM = mean_ci log(x), vectorized 32B LDS reads ----
  for (int e = tid; e < LM_FLOATS; e += 256) {
    int xw = e % DIM;
    int dy = (e / DIM) % TILE;
    int dz = e / (DIM*TILE);
    const v16h v = *(const v16h*)&lx[(((dz+1)*HALO + (dy+1))*WROW + (xw+1))*CIN];
    float s = 0.0f;
    #pragma unroll
    for (int ci = 0; ci < CIN; ++ci) s += (float)v[ci];
    logM[e] = s * (1.0f/16.0f);
  }
  __syncthreads();

  // ---- Implicit-GEMM conv: D[co, vox] += W_pair[co,K] * LogX_pair[K, vox] ----
  const int wave  = tid >> 5;
  const int lane  = tid & 31;
  const int nsel  = lane & 15;   // N (voxel within segment)
  const int khalf = lane >> 4;   // which tap of the pair this lane's B rows cover

  // Pair-invariant per-segment LDS base offsets (halves) — computed once.
  int segbase[12];
  #pragma unroll
  for (int s = 0; s < 12; ++s) {
    const int seg = wave*12 + s;
    const int ws  = seg % 6;
    const int p   = seg / 6;
    const int dy  = p & 3;
    const int dz  = p >> 2;
    segbase[s] = ((dz*HALO + dy)*WROW + (ws*16 + nsel))*CIN;
  }

  v8f acc[12];
  v8f zero = {};
  #pragma unroll
  for (int s = 0; s < 12; ++s) acc[s] = zero;

  for (int pair = 0; pair < NPAIR; ++pair) {
    const v16h afrag = *(const v16h*)&wlds[(pair*32 + lane)*16];
    int t = pair*2 + khalf;
    t = (t > 26) ? 0 : t;                 // padded tap: weights are zero anyway
    const int oz = t / 9;
    const int oy = (t / 3) % 3;
    const int ox = t - (t / 3) * 3;       // offsets 0..2 (halo +1 and tap -1 cancel)
    const int delta = ((oz*HALO + oy)*WROW + ox)*CIN;   // single add per segment below
    #pragma unroll
    for (int s = 0; s < 12; ++s) {
      const v16h bfrag = *(const v16h*)&lx[segbase[s] + delta];
      acc[s] = __builtin_amdgcn_wmma_f32_16x16x32_f16(
                   false, afrag, false, bfrag, (short)0, acc[s], false, false);
    }
  }

  // ---- Epilogue: out = exp(conv + (1 - wsum[co]) * logM) ----
  #pragma unroll
  for (int s = 0; s < 12; ++s) {
    const int seg = wave*12 + s;
    const int ws  = seg % 6;
    const int p   = seg / 6;
    const int dy  = p & 3;
    const int dz  = p >> 2;
    const int xw  = ws*16 + nsel;
    const float lm = logM[(dz*TILE + dy)*DIM + xw];
    const int gz = zt*TILE + dz;
    const int gy = yt*TILE + dy;
    #pragma unroll
    for (int r = 0; r < 8; ++r) {
      const int co = khalf*8 + r;         // C/D layout: lanes>=16 hold M = r+8
      const float v = acc[s][r] + wadj[co]*lm;
      out[(((b*COUT + co)*DIM + gz)*DIM + gy)*DIM + xw] = __expf(v);
    }
  }
}

extern "C" void kernel_launch(void* const* d_in, const int* in_sizes, int n_in,
                              void* d_out, int out_size, void* d_ws, size_t ws_size,
                              hipStream_t stream) {
  const float* x = (const float*)d_in[0];   // [2,16,96,96,96] f32
  const float* w = (const float*)d_in[1];   // [16,3,3,3,16]   f32
  float* out = (float*)d_out;               // [2,16,96,96,96] f32
  dim3 grid(DIM/TILE, DIM/TILE, 2);         // (yt, zt, b)
  mvc_wmma_kernel<<<grid, 256, LDS_BYTES, stream>>>(x, w, out);
}
